// GraphConvHighWayLayer_77043123356300
// MI455X (gfx1250) — matrix operand order repaired and checked by reading, not compiled, not run on hardware
//
#include <hip/hip_runtime.h>
#include <hip/hip_bf16.h>
#include <math.h>

// ---------------------------------------------------------------------------
// GraphConv + HighWay layer for MI455X (gfx1250, wave32, WMMA).
//  F=3 fields, N nodes, D=O=256, E edges.
//  Pipeline:
//   1. deg init (=2.0 self-loop) + atomic accumulate edge weights
//   2. dinv = rsqrt(deg)
//   3. norm[e] = dinv[row]*w*dinv[col]
//   4. x -> bf16 (single pass), build WT bf16 [F][768][256] (Wg^T|Wr|Wh^T)
//   5. WMMA bf16 GEMM: xg / res / sigmoid-gate  ([F][N][256] fp32 each)
//   6. zero d_out, scatter-add out[f][col] += xg[f][row]*norm (HW f32 atomics)
//   7. combine: gate*(gcn + 2*dinv^2*xg) + (1-gate)*res, leaky relu
// ---------------------------------------------------------------------------

typedef __bf16 bf16x16 __attribute__((ext_vector_type(16)));
typedef __bf16 bf16x8  __attribute__((ext_vector_type(8)));
typedef float  f32x8   __attribute__((ext_vector_type(8)));

#define DIMK   256
#define JOUT   768
#define NFIELD 3
#define NEG_SLOPE 0.01f
#define EPB 8   // edges per block in scatter

__device__ __forceinline__ int imin(int a, int b) { return a < b ? a : b; }

// Hardware f32 atomic add, no return (STOREcnt path, L2 atomic units).
__device__ __forceinline__ void gatomic_add_f32(float* p, float v) {
  asm volatile("global_atomic_add_f32 %0, %1, off" :: "v"(p), "v"(v) : "memory");
}

// ------------------------------ small kernels ------------------------------

__global__ void gchw_init_deg(float* __restrict__ deg, int n) {
  int i = blockIdx.x * blockDim.x + threadIdx.x;
  if (i < n) deg[i] = 2.0f;   // improved GCN self-loop weight
}

__global__ void gchw_accum_deg(const int* __restrict__ col,
                               const float* __restrict__ w,
                               float* __restrict__ deg, int e) {
  int i = blockIdx.x * blockDim.x + threadIdx.x;
  if (i < e) gatomic_add_f32(deg + col[i], w[i]);
}

__global__ void gchw_dinv(const float* __restrict__ deg,
                          float* __restrict__ dinv, int n) {
  int i = blockIdx.x * blockDim.x + threadIdx.x;
  if (i < n) {
    float d = deg[i];
    dinv[i] = (d > 0.0f) ? rsqrtf(d) : 0.0f;
  }
}

__global__ void gchw_norm(const int* __restrict__ row, const int* __restrict__ col,
                          const float* __restrict__ w, const float* __restrict__ dinv,
                          float* __restrict__ norm, int e) {
  int i = blockIdx.x * blockDim.x + threadIdx.x;
  if (i < e) norm[i] = dinv[row[i]] * w[i] * dinv[col[i]];
}

// One-pass fp32 -> bf16 (8 elems/thread: b128 load, 4x cvt_pk, b128 store)
__global__ void gchw_cvt_x(const float* __restrict__ x, __bf16* __restrict__ xb,
                           long long n8) {
  long long i = (long long)blockIdx.x * blockDim.x + threadIdx.x;
  if (i >= n8) return;
  const float4* p = reinterpret_cast<const float4*>(x) + i * 2;
  float4 u = p[0];
  float4 v = p[1];
  bf16x8 o;
  o[0] = (__bf16)u.x; o[1] = (__bf16)u.y; o[2] = (__bf16)u.z; o[3] = (__bf16)u.w;
  o[4] = (__bf16)v.x; o[5] = (__bf16)v.y; o[6] = (__bf16)v.z; o[7] = (__bf16)v.w;
  reinterpret_cast<bf16x8*>(xb)[i] = o;
}

// WT[f][jo][d]: jo<256 -> Wg[f][d][jo]; jo<512 -> Wr[f][jo-256][d]; else Wh[f][d][jo-512]
__global__ void gchw_build_wt(const float* __restrict__ Wg,
                              const float* __restrict__ Wr,
                              const float* __restrict__ Wh,
                              __bf16* __restrict__ WT) {
  int idx = blockIdx.x * blockDim.x + threadIdx.x;
  if (idx >= NFIELD * JOUT * DIMK) return;
  int d  = idx & 255;
  int jo = (idx >> 8) % JOUT;
  int f  = idx / (JOUT * DIMK);
  float v;
  if (jo < 256)      v = Wg[((size_t)f * 256 + d) * 256 + jo];
  else if (jo < 512) v = Wr[((size_t)f * 256 + (jo - 256)) * 256 + d];
  else               v = Wh[((size_t)f * 256 + d) * 256 + (jo - 512)];
  WT[idx] = (__bf16)v;
}

__global__ void gchw_zero4(float4* __restrict__ p, int n4) {
  int i = blockIdx.x * blockDim.x + threadIdx.x;
  if (i < n4) p[i] = make_float4(0.f, 0.f, 0.f, 0.f);
}

// ------------------------------ WMMA GEMM ----------------------------------
// Block: 256 thr = 8 waves (2 x 4); block tile 64(M) x 256(out-chunk).
// grid = ((N+63)/64, 3 jo-chunks, 3 fields). Wave tile 32x64 = 2x4 accums.

template<bool GUARD>
__device__ __forceinline__ void gchw_store_tile(
    float* __restrict__ dst, const f32x8 acc[2][4],
    int waveM, int waveN, int half, int l16, int which, int Nn) {
#pragma unroll
  for (int i = 0; i < 2; ++i) {
    const int mbase = waveM + i * 16 + half * 8;   // C/D: VGPR r -> M = r (+8 hi half)
#pragma unroll
    for (int j = 0; j < 4; ++j) {
      const int o = waveN + j * 16 + l16;
#pragma unroll
      for (int r = 0; r < 8; ++r) {
        const int m = mbase + r;
        if (!GUARD || m < Nn) {
          float v = acc[i][j][r];
          if (which == 2) v = 1.0f / (1.0f + __expf(-v));   // fused sigmoid
          dst[(size_t)m * DIMK + o] = v;
        }
      }
    }
  }
}

__global__ __launch_bounds__(256)
void gchw_gemm_wmma(const __bf16* __restrict__ xb,    // [F][N][256] bf16
                    const __bf16* __restrict__ WT,    // [F][768][256] bf16
                    float* __restrict__ xg,
                    float* __restrict__ res,
                    float* __restrict__ gate,
                    int Nn) {
  const int f      = blockIdx.z;
  const int which  = blockIdx.y;            // 0:xg 1:res 2:gate
  const int blockM = blockIdx.x * 64;
  const int lane   = threadIdx.x & 31;
  const int wid    = threadIdx.x >> 5;
  const int half   = lane >> 4;             // 0 | 1
  const int l16    = lane & 15;
  const int waveM  = blockM + (wid >> 2) * 32;
  const int waveN  = (wid & 3) * 64;        // within 256-chunk

  const __bf16* xf = xb + (size_t)f * Nn * DIMK;
  const __bf16* wf = WT + ((size_t)f * JOUT + (size_t)which * 256) * DIMK;

  f32x8 acc[2][4];
#pragma unroll
  for (int i = 0; i < 2; ++i)
#pragma unroll
    for (int j = 0; j < 4; ++j)
      acc[i][j] = (f32x8){0.f, 0.f, 0.f, 0.f, 0.f, 0.f, 0.f, 0.f};

  const __bf16* arow0 = xf + (size_t)imin(waveM + l16,      Nn - 1) * DIMK;
  const __bf16* arow1 = xf + (size_t)imin(waveM + 16 + l16, Nn - 1) * DIMK;
  const __bf16* brow[4];
#pragma unroll
  for (int j = 0; j < 4; ++j)
    brow[j] = wf + (size_t)(waveN + j * 16 + l16) * DIMK;

  for (int k = 0; k < DIMK; k += 32) {
    bf16x16 a[2], b[4];
    // A 16x32 bf16 frag: lane<16: K[k..k+15]; lane>=16: K[k+16..k+31] (matches B)
    {
      a[0] = *reinterpret_cast<const bf16x16*>(arow0 + k + half * 16);
      a[1] = *reinterpret_cast<const bf16x16*>(arow1 + k + half * 16);
    }
    // B 32x16 frag: lane holds column n=l16, K[k..k+15] (lo half-wave) or
    // K[k+16..k+31] (hi half-wave); contiguous in B^T storage.
#pragma unroll
    for (int j = 0; j < 4; ++j)
      b[j] = *reinterpret_cast<const bf16x16*>(brow[j] + k + half * 16);
#pragma unroll
    for (int i = 0; i < 2; ++i)
#pragma unroll
      for (int j = 0; j < 4; ++j)
        acc[i][j] = __builtin_amdgcn_wmma_f32_16x16x32_bf16(
            false, a[i], false, b[j], (short)0, acc[i][j], false, false);
  }

  float* dst = (which == 0) ? xg : (which == 1) ? res : gate;
  dst += (size_t)f * Nn * DIMK;
  if (blockM + 64 <= Nn)
    gchw_store_tile<false>(dst, acc, waveM, waveN, half, l16, which, Nn);
  else
    gchw_store_tile<true >(dst, acc, waveM, waveN, half, l16, which, Nn);
}

// ------------------------------ scatter ------------------------------------
// One 256-thread block handles EPB edges for one field; each thread owns one
// of the 256 features: coalesced 1KB gather + 1KB of HW f32 atomics per edge.
// xg (61MB) is L2-resident (192MB L2), so random gathers mostly hit L2.

__global__ __launch_bounds__(256)
void gchw_scatter(const int* __restrict__ row, const int* __restrict__ col,
                  const float* __restrict__ norm, const float* __restrict__ xg,
                  float* __restrict__ out, int E, int Nn) {
  const int f = blockIdx.y;
  const int o = threadIdx.x;
  const float* xgf = xg  + (size_t)f * Nn * DIMK;
  float*       of  = out + (size_t)f * Nn * DIMK;
  const int base = blockIdx.x * EPB;
#pragma unroll 1
  for (int i = 0; i < EPB; ++i) {
    const int e = base + i;
    if (e >= E) break;
    const int   r  = row[e];
    const int   c  = col[e];
    const float nv = norm[e];
    if (e + 1 < E)   // warm next gather line (global_prefetch_b8)
      __builtin_prefetch(xgf + (size_t)row[e + 1] * DIMK + o, 0, 1);
    const float v = xgf[(size_t)r * DIMK + o] * nv;
    gatomic_add_f32(of + (size_t)c * DIMK + o, v);
  }
}

// ------------------------------ combine ------------------------------------
// rep = gate*(gcn_accum + 2*dinv^2*xg) + (1-gate)*res ; leaky_relu

__global__ void gchw_combine(float* __restrict__ out, const float* __restrict__ xg,
                             const float* __restrict__ res, const float* __restrict__ gate,
                             const float* __restrict__ dinv, int Nn, int n4) {
  int i = blockIdx.x * blockDim.x + threadIdx.x;
  if (i >= n4) return;
  const size_t base = (size_t)i * 4;
  const int n = (int)((base >> 8) % (size_t)Nn);
  const float dn = dinv[n];
  const float sn = 2.0f * dn * dn;
  float4 a  = reinterpret_cast<float4*>(out)[i];
  float4 xv = reinterpret_cast<const float4*>(xg)[i];
  float4 g  = reinterpret_cast<const float4*>(gate)[i];
  float4 r  = reinterpret_cast<const float4*>(res)[i];
  float4 rep;
  rep.x = g.x * (a.x + sn * xv.x) + (1.0f - g.x) * r.x;
  rep.y = g.y * (a.y + sn * xv.y) + (1.0f - g.y) * r.y;
  rep.z = g.z * (a.z + sn * xv.z) + (1.0f - g.z) * r.z;
  rep.w = g.w * (a.w + sn * xv.w) + (1.0f - g.w) * r.w;
  rep.x = rep.x >= 0.f ? rep.x : NEG_SLOPE * rep.x;
  rep.y = rep.y >= 0.f ? rep.y : NEG_SLOPE * rep.y;
  rep.z = rep.z >= 0.f ? rep.z : NEG_SLOPE * rep.z;
  rep.w = rep.w >= 0.f ? rep.w : NEG_SLOPE * rep.w;
  reinterpret_cast<float4*>(out)[i] = rep;
}

// ------------------------------ launch -------------------------------------

static inline size_t au256(size_t v) { return (v + 255) & ~(size_t)255; }

extern "C" void kernel_launch(void* const* d_in, const int* in_sizes, int n_in,
                              void* d_out, int out_size, void* d_ws, size_t ws_size,
                              hipStream_t stream) {
  const float* x  = (const float*)d_in[0];   // [3,N,256]
  const int*   ei = (const int*)d_in[1];     // [2,E]
  const float* ew = (const float*)d_in[2];   // [E]
  const float* Wg = (const float*)d_in[3];   // [3,256,256]
  const float* Wr = (const float*)d_in[4];   // [3,256,256] ([F,O,D])
  const float* Wh = (const float*)d_in[5];   // [3,256,256]
  float* out = (float*)d_out;

  const int E  = in_sizes[2];
  const int Nn = in_sizes[0] / (NFIELD * DIMK);
  const int* row = ei;
  const int* col = ei + E;

  // Workspace carve-up (~218 MB): deg, dinv, norm, WT, x_bf16, xg, res, gate
  char* wsp = (char*)d_ws;
  size_t off = 0;
  float*  deg  = (float*)(wsp + off);  off = au256(off + (size_t)Nn * 4);
  float*  dinv = (float*)(wsp + off);  off = au256(off + (size_t)Nn * 4);
  float*  norm = (float*)(wsp + off);  off = au256(off + (size_t)E * 4);
  __bf16* WT   = (__bf16*)(wsp + off); off = au256(off + (size_t)NFIELD * JOUT * DIMK * 2);
  __bf16* xb   = (__bf16*)(wsp + off); off = au256(off + (size_t)NFIELD * Nn * DIMK * 2);
  const size_t tensor_bytes = (size_t)NFIELD * Nn * DIMK * 4;
  float*  xg   = (float*)(wsp + off);  off = au256(off + tensor_bytes);
  float*  res  = (float*)(wsp + off);  off = au256(off + tensor_bytes);
  float*  gate = (float*)(wsp + off);  off = au256(off + tensor_bytes);
  (void)ws_size; (void)n_in;

  const int total = NFIELD * Nn * DIMK;
  const int n4 = total / 4;
  const long long n8 = (long long)total / 8;

  // 1-3: GCN symmetric normalization
  gchw_init_deg <<<(Nn + 255) / 256, 256, 0, stream>>>(deg, Nn);
  gchw_accum_deg<<<(E  + 255) / 256, 256, 0, stream>>>(col, ew, deg, E);
  gchw_dinv     <<<(Nn + 255) / 256, 256, 0, stream>>>(deg, dinv, Nn);
  gchw_norm     <<<(E  + 255) / 256, 256, 0, stream>>>(row, col, ew, dinv, norm, E);

  // 4: one-pass bf16 conversion of x + packed transposed weights
  gchw_cvt_x    <<<(int)((n8 + 255) / 256), 256, 0, stream>>>(x, xb, n8);
  gchw_build_wt <<<(NFIELD * JOUT * DIMK + 255) / 256, 256, 0, stream>>>(Wg, Wr, Wh, WT);

  // 5: WMMA GEMMs (xg | res | sigmoid-gate)
  dim3 ggrid((Nn + 63) / 64, 3, NFIELD);
  gchw_gemm_wmma<<<ggrid, 256, 0, stream>>>(xb, WT, xg, res, gate, Nn);

  // 6: zero accumulator (d_out) then edge scatter with HW f32 atomics
  gchw_zero4<<<(n4 + 255) / 256, 256, 0, stream>>>((float4*)out, n4);
  dim3 sgrid((E + EPB - 1) / EPB, NFIELD);
  gchw_scatter<<<sgrid, 256, 0, stream>>>(row, col, norm, xg, out, E, Nn);

  // 7: highway combine + leaky relu (self-loop folded in analytically)
  gchw_combine<<<(n4 + 255) / 256, 256, 0, stream>>>(out, xg, res, gate, dinv, Nn, n4);
}
